// KAN_PROSPECT_ADR_59502476919296
// MI455X (gfx1250) — matrix operand order repaired and checked by reading, not compiled, not run on hardware
//
#include <hip/hip_runtime.h>
#include <math.h>

// ---------------------------------------------------------------------------
// Types for gfx1250 WMMA (wave32). bf16 data, f32 accumulate.
// ---------------------------------------------------------------------------
typedef __attribute__((ext_vector_type(16))) __bf16 bf16x16;
typedef __attribute__((ext_vector_type(8)))  float  f32x8;

#define DEVFN __device__ __forceinline__

DEVFN unsigned short f2bf(float f) {           // round-to-nearest-even f32 -> bf16
  unsigned u = __float_as_uint(f);
  u += 0x7FFFu + ((u >> 16) & 1u);
  return (unsigned short)(u >> 16);
}

// order-preserving float<->uint encoding for atomicMax on floats
DEVFN unsigned f2ord(float f) {
  int i = __float_as_int(f);
  unsigned u = (unsigned)i;
  return (i >= 0) ? (u | 0x80000000u) : ~u;
}
DEVFN float ord2f(unsigned k) {
  unsigned u = (k & 0x80000000u) ? (k & 0x7FFFFFFFu) : ~k;
  return __uint_as_float(u);
}

DEVFN void edge_sd(long long e, const int* ei, int E, int Nn, int& src, int& dst) {
  if (e < E) { src = ei[e]; dst = ei[(long long)E + e]; }
  else       { src = dst = (int)(e - E); }
}

// CDNA5 async global->LDS DMA (ASYNCcnt). LDS target is the low 32 bits of the
// generic pointer (LDS aperture keeps the wave-relative offset in addr[31:0]).
DEVFN void async_ld_b128(unsigned lds_off, const unsigned short* g) {
  asm volatile("global_load_async_to_lds_b128 %0, %1, off"
               :: "v"(lds_off), "v"(g) : "memory");
}
DEVFN void wait_async0() { asm volatile("s_wait_asynccnt 0" ::: "memory"); }

// ---------------------------------------------------------------------------
// Utility kernels
// ---------------------------------------------------------------------------
__global__ void k_fill_f32(float* p, float v, long long n) {
  long long i = (long long)blockIdx.x * blockDim.x + threadIdx.x;
  if (i < n) p[i] = v;
}
__global__ void k_fill_u32(unsigned* p, unsigned v, long long n) {
  long long i = (long long)blockIdx.x * blockDim.x + threadIdx.x;
  if (i < n) p[i] = v;
}

// f32 (rows x K, stride Ss) -> bf16 (rows x Kp) zero-padded; optional SiLU
__global__ void k_cvt_bf16(const float* __restrict__ src, int Ss,
                           unsigned short* __restrict__ dst, int Kp, int K,
                           long long rows, int do_silu) {
  long long i = (long long)blockIdx.x * blockDim.x + threadIdx.x;
  if (i >= rows * (long long)Kp) return;
  int k = (int)(i % Kp);
  long long r = i / Kp;
  float v = 0.f;
  if (k < K) {
    v = src[r * (long long)Ss + k];
    if (do_silu) v = v / (1.f + __expf(-v));
  }
  dst[i] = f2bf(v);
}

// pack (K x Nout) f32 ("in,out" layout, used as x@W) -> bf16 (Npad x Kp) transposed
__global__ void k_pack_T(const float* __restrict__ Wsrc, int K, int Nout,
                         unsigned short* __restrict__ Wt, int Kp, int Npad) {
  long long i = (long long)blockIdx.x * blockDim.x + threadIdx.x;
  if (i >= (long long)Npad * Kp) return;
  int k = (int)(i % Kp);
  int n = (int)(i / Kp);
  float v = (n < Nout && k < K) ? Wsrc[(long long)k * Nout + n] : 0.f;
  Wt[i] = f2bf(v);
}

// pack (Nout x K) f32 ("out,in" layout, KAN base) -> bf16 (Npad x Kp)
__global__ void k_pack_R(const float* __restrict__ Wsrc, int K, int Nout,
                         unsigned short* __restrict__ Wt, int Kp, int Npad) {
  long long i = (long long)blockIdx.x * blockDim.x + threadIdx.x;
  if (i >= (long long)Npad * Kp) return;
  int k = (int)(i % Kp);
  int n = (int)(i / Kp);
  float v = (n < Nout && k < K) ? Wsrc[(long long)n * K + k] : 0.f;
  Wt[i] = f2bf(v);
}

// pack spline (Nout x In x 12) * scaler (Nout x In) -> bf16 (Npad x In*12)
__global__ void k_pack_spline(const float* __restrict__ sp, const float* __restrict__ sc,
                              int In, int Nout, unsigned short* __restrict__ Wt, int Npad) {
  int Kp = In * 12;
  long long i = (long long)blockIdx.x * blockDim.x + threadIdx.x;
  if (i >= (long long)Npad * Kp) return;
  int k = (int)(i % Kp);
  int n = (int)(i / Kp);
  float v = 0.f;
  if (n < Nout) {
    int ii = k / 12, t = k % 12;
    v = sp[((long long)n * In + ii) * 12 + t] * sc[(long long)n * In + ii];
  }
  Wt[i] = f2bf(v);
}

// ---------------------------------------------------------------------------
// WMMA GEMM: C(MxNs f32) = A(MxKp bf16) * W(Npad x Kp bf16)^T  [+ C if beta]
// Requirements: M % 512 == 0 (true for all call sites: 40960, 1024),
//               Kp % 64 == 0.
// Block = 8 waves; each wave owns 4 16x16 C tiles. The 16x64 W slab is
// double-buffered in LDS via global_load_async_to_lds_b128 (ASYNCcnt), so the
// DMA overlaps the 8 straight-line WMMAs each wave issues per stage.
// No predication in the hot loop -> no exec juggling, accs stay in tied regs.
// Fragment layouts per CDNA5 ISA 7.12.2.
// ---------------------------------------------------------------------------
#define GEMM_MT 4

__global__ void k_gemm_wmma(const unsigned short* __restrict__ A,
                            const unsigned short* __restrict__ W,
                            float* __restrict__ C,
                            int M, int Kp, int Ns, int beta) {
  (void)M;
  __shared__ __align__(16) unsigned short ldsW[2][16 * 64];  // 2 x 2KB ping-pong
  const int wave = threadIdx.x >> 5;
  const int lane = threadIdx.x & 31;
  const int ntile = blockIdx.x;
  const int half = lane >> 4;          // 0/1
  const int l15  = lane & 15;
  const int kbA = half * 8;            // A: lanes16-31 hold K 8-15 / 24-31
  const int kbB = half * 8;            // B: lanes16-31 hold K 8-15 / 24-31
  const int t = threadIdx.x;

  const int mt0 = (blockIdx.y * 8 + wave) * GEMM_MT;
  const int n0 = ntile * 16;

  f32x8 acc[GEMM_MT];
  const unsigned short* Ar[GEMM_MT];
#pragma unroll
  for (int m = 0; m < GEMM_MT; ++m) {
    Ar[m] = A + (long long)((mt0 + m) * 16 + l15) * Kp;
    if (beta) {
      const int m0 = (mt0 + m) * 16;
#pragma unroll
      for (int v = 0; v < 8; ++v)
        acc[m][v] = C[(long long)(m0 + v + 8 * half) * Ns + n0 + l15];
    } else {
#pragma unroll
      for (int v = 0; v < 8; ++v) acc[m][v] = 0.f;
    }
  }

  // DMA producers: threads 0..127 (waves 0-3) each move 16B of the 16x64 slab
  const bool loader = (t < 128);                       // wave-uniform
  const unsigned short* wsrc =
      W + (long long)(ntile * 16 + (t >> 3)) * Kp + (t & 7) * 8;
  const unsigned ldsbase = (unsigned)(unsigned long long)&ldsW[0][0];
  const unsigned myslot = (unsigned)t * 16u;

  if (loader) async_ld_b128(ldsbase + myslot, wsrc);   // preload stage 0
  wait_async0();
  __syncthreads();

  union Frag { bf16x16 v; uint4 q[2]; };

  int buf = 0;
  for (int k0 = 0; k0 < Kp; k0 += 64) {
    // kick DMA for the next stage into the other buffer, overlapped with WMMA
    if (loader && (k0 + 64 < Kp))
      async_ld_b128(ldsbase + (unsigned)((buf ^ 1) * 2048) + myslot,
                    wsrc + k0 + 64);

    // gather the whole stage's fragments first (one wait covers all loads)...
    Frag fb[2];
#pragma unroll
    for (int j = 0; j < 2; ++j) {
      const unsigned short* wp = &ldsW[buf][l15 * 64 + j * 32 + kbB];
      fb[j].q[0] = *(const uint4*)(wp);
      fb[j].q[1] = *(const uint4*)(wp + 16);
    }
    Frag fa[GEMM_MT][2];
#pragma unroll
    for (int m = 0; m < GEMM_MT; ++m) {
#pragma unroll
      for (int j = 0; j < 2; ++j) {
        fa[m][j].q[0] = *(const uint4*)(Ar[m] + k0 + j * 32 + kbA);
        fa[m][j].q[1] = *(const uint4*)(Ar[m] + k0 + j * 32 + 16 + kbA);
      }
    }
    // ...then issue the 8 WMMAs back-to-back into tied accumulators
#pragma unroll
    for (int j = 0; j < 2; ++j) {
#pragma unroll
      for (int m = 0; m < GEMM_MT; ++m) {
        acc[m] = __builtin_amdgcn_wmma_f32_16x16x32_bf16(
            false, fa[m][j].v, false, fb[j].v, (short)0, acc[m], false, false);
      }
    }

    wait_async0();          // my DMA for next stage has landed
    __syncthreads();        // everyone done reading buf / writing buf^1
    buf ^= 1;
  }

#pragma unroll
  for (int m = 0; m < GEMM_MT; ++m) {
    const int m0 = (mt0 + m) * 16;
#pragma unroll
    for (int v = 0; v < 8; ++v)
      C[(long long)(m0 + v + 8 * half) * Ns + n0 + l15] = acc[m][v];
  }
}

// ---------------------------------------------------------------------------
// GAT / GCN graph kernels
// ---------------------------------------------------------------------------
__global__ void k_gat_scores(const float* __restrict__ xl, int Ns,
                             const float* __restrict__ av, float* __restrict__ s,
                             int Nn, int H, int C) {
  long long i = (long long)blockIdx.x * blockDim.x + threadIdx.x;
  if (i >= (long long)Nn * H) return;
  int n = (int)(i / H), h = (int)(i % H);
  const float* xp = xl + (long long)n * Ns + h * C;
  const float* ap = av + h * C;
  float acc = 0.f;
  for (int c = 0; c < C; ++c) acc += xp[c] * ap[c];
  s[i] = acc;
}

__global__ void k_edge_max(const int* __restrict__ ei, int E, int Nn, int H,
                           const float* __restrict__ ss, const float* __restrict__ sd,
                           unsigned* __restrict__ m) {
  long long t = (long long)blockIdx.x * blockDim.x + threadIdx.x;
  if (t >= (long long)(E + Nn) * H) return;
  int h = (int)(t % H); long long e = t / H;
  int src, dst; edge_sd(e, ei, E, Nn, src, dst);
  float v = ss[(long long)src * H + h] + sd[(long long)dst * H + h];
  v = (v > 0.f) ? v : 0.2f * v;
  atomicMax(&m[(long long)dst * H + h], f2ord(v));
}

__global__ void k_edge_expsum(const int* __restrict__ ei, int E, int Nn, int H,
                              const float* __restrict__ ss, const float* __restrict__ sd,
                              const unsigned* __restrict__ m,
                              float* __restrict__ wbuf, float* __restrict__ ssum) {
  long long t = (long long)blockIdx.x * blockDim.x + threadIdx.x;
  if (t >= (long long)(E + Nn) * H) return;
  int h = (int)(t % H); long long e = t / H;
  int src, dst; edge_sd(e, ei, E, Nn, src, dst);
  float v = ss[(long long)src * H + h] + sd[(long long)dst * H + h];
  v = (v > 0.f) ? v : 0.2f * v;
  float w = __expf(v - ord2f(m[(long long)dst * H + h]));
  wbuf[t] = w;
  atomicAdd(&ssum[(long long)dst * H + h], w);
}

__global__ void k_edge_aggr(const int* __restrict__ ei, int E, int Nn, int H, int C, int Ns,
                            const float* __restrict__ xl, const float* __restrict__ wbuf,
                            const float* __restrict__ ssum, float* __restrict__ out) {
  long long t = (long long)blockIdx.x * blockDim.x + threadIdx.x;
  if (t >= (long long)(E + Nn) * H) return;
  int h = (int)(t % H); long long e = t / H;
  int src, dst; edge_sd(e, ei, E, Nn, src, dst);
  float coef = wbuf[t] / ssum[(long long)dst * H + h];
  const float* xp = xl + (long long)src * Ns + h * C;
  float* op = out + (long long)dst * Ns + h * C;
  for (int c = 0; c < C; ++c) atomicAdd(&op[c], coef * xp[c]);
}

__global__ void k_deg(const int* __restrict__ ei, int E, int Nn, float* __restrict__ deg) {
  long long e = (long long)blockIdx.x * blockDim.x + threadIdx.x;
  if (e >= (long long)(E + Nn)) return;
  int src, dst; edge_sd(e, ei, E, Nn, src, dst);
  atomicAdd(&deg[dst], 1.f);
}

__global__ void k_gcn_aggr(const int* __restrict__ ei, int E, int Nn,
                           const float* __restrict__ deg, const float* __restrict__ xw,
                           int Ns, float* __restrict__ out, int F, int CH) {
  int nch = F / CH;
  long long t = (long long)blockIdx.x * blockDim.x + threadIdx.x;
  if (t >= (long long)(E + Nn) * nch) return;
  int ch = (int)(t % nch); long long e = t / nch;
  int src, dst; edge_sd(e, ei, E, Nn, src, dst);
  float nrm = rsqrtf(deg[src] * deg[dst]);
  const float* xp = xw + (long long)src * Ns + ch * CH;
  float* op = out + (long long)dst * Ns + ch * CH;
  for (int c = 0; c < CH; ++c) atomicAdd(&op[c], nrm * xp[c]);
}

// in-place epilogue: (+bias) -> (elu) -> (affine g,b) -> (relu)
__global__ void k_epilogue(float* __restrict__ x, int Ns, long long rows, int F,
                           const float* __restrict__ bias, int do_elu,
                           const float* __restrict__ g, const float* __restrict__ bb,
                           int do_relu) {
  long long i = (long long)blockIdx.x * blockDim.x + threadIdx.x;
  if (i >= rows * F) return;
  long long r = i / F; int j = (int)(i % F);
  float v = x[r * (long long)Ns + j];
  if (bias) v += bias[j];
  if (do_elu) v = (v > 0.f) ? v : expm1f(v);
  if (g) v = v * g[j] + bb[j];
  if (do_relu) v = fmaxf(v, 0.f);
  x[r * (long long)Ns + j] = v;
}

__global__ void k_pool_max(const float* __restrict__ h, int Ns, float* __restrict__ g,
                           int Bn, int F, int npg) {
  long long i = (long long)blockIdx.x * blockDim.x + threadIdx.x;
  if (i >= (long long)Bn * F) return;
  int b = (int)(i / F), j = (int)(i % F);
  float m = -3.402823466e38f;
  for (int t = 0; t < npg; ++t)
    m = fmaxf(m, h[(long long)(b * npg + t) * Ns + j]);
  g[i] = m;
}

// ---------------------------------------------------------------------------
// KAN B-spline basis (GRID=10, K=2 -> 12 quadratic bases over [-1,1]),
// emitted directly as bf16 with layout (rows x In*12) so the spline branch
// becomes one WMMA GEMM with K = In*12.
// ---------------------------------------------------------------------------
__global__ void k_bspline_bf16(const float* __restrict__ X, int Ss, int In,
                               unsigned short* __restrict__ dst, long long rows) {
  long long i = (long long)blockIdx.x * blockDim.x + threadIdx.x;
  if (i >= rows * In) return;
  long long r = i / In; int c = (int)(i % In);
  float x = X[r * (long long)Ss + c];
  const float h = 0.2f;
  float b0[14];
#pragma unroll
  for (int t = 0; t < 14; ++t) {
    float gl = (t - 2) * h - 1.f, gr = (t - 1) * h - 1.f;
    b0[t] = (x >= gl && x < gr) ? 1.f : 0.f;
  }
  float b1[13];
#pragma unroll
  for (int t = 0; t < 13; ++t) {
    float g0 = (t - 2) * h - 1.f, g1 = (t - 1) * h - 1.f, g2 = t * h - 1.f;
    b1[t] = (x - g0) / (g1 - g0) * b0[t] + (g2 - x) / (g2 - g1) * b0[t + 1];
  }
  float b2v[12];
#pragma unroll
  for (int t = 0; t < 12; ++t) {
    float g0 = (t - 2) * h - 1.f, g1 = (t - 1) * h - 1.f;
    float g2 = t * h - 1.f, g3 = (t + 1) * h - 1.f;
    b2v[t] = (x - g0) / (g2 - g0) * b1[t] + (g3 - x) / (g3 - g1) * b1[t + 1];
  }
  unsigned short* dp = dst + r * (long long)(In * 12) + (long long)c * 12;
#pragma unroll
  for (int t = 0; t < 12; ++t) dp[t] = f2bf(b2v[t]);
}

// ---------------------------------------------------------------------------
// Token branch: embedding gather + Conv1d(16->32, k=8) fused with BN + ReLU,
// written directly in (B, f*121+l) order for the following KAN.
// ---------------------------------------------------------------------------
__global__ void k_emb(const float* __restrict__ emb, const int* __restrict__ tgt,
                      float* __restrict__ out, int Bn) {
  long long i = (long long)blockIdx.x * blockDim.x + threadIdx.x;
  if (i >= (long long)Bn * 16 * 128) return;
  int l = (int)(i % 128);
  long long bi = i / 128;
  int b = (int)(bi / 16), c = (int)(bi % 16);
  out[i] = emb[(long long)tgt[b * 16 + c] * 128 + l];
}

__global__ void k_conv(const float* __restrict__ ex, const float* __restrict__ cw,
                       const float* __restrict__ cb, const float* __restrict__ g,
                       const float* __restrict__ bb, float* __restrict__ out, int Bn) {
  long long i = (long long)blockIdx.x * blockDim.x + threadIdx.x;
  if (i >= (long long)Bn * 32 * 121) return;
  int l = (int)(i % 121);
  long long t2 = i / 121;
  int f = (int)(t2 % 32), b = (int)(t2 / 32);
  float acc = cb[f];
  for (int c = 0; c < 16; ++c) {
    const float* ep = ex + ((long long)b * 16 + c) * 128 + l;
    const float* wp = cw + ((long long)f * 16 + c) * 8;
#pragma unroll
    for (int t = 0; t < 8; ++t) acc += ep[t] * wp[t];
  }
  acc = acc * g[f] + bb[f];
  out[(long long)b * 3872 + f * 121 + l] = fmaxf(acc, 0.f);
}

__global__ void k_copycols(const float* __restrict__ src, int Ss,
                           float* __restrict__ dst, int Ds, int dcol, int F,
                           long long rows) {
  long long i = (long long)blockIdx.x * blockDim.x + threadIdx.x;
  if (i >= rows * F) return;
  long long r = i / F; int j = (int)(i % F);
  dst[r * (long long)Ds + dcol + j] = src[r * (long long)Ss + j];
}

// ---------------------------------------------------------------------------
// Host orchestration
// ---------------------------------------------------------------------------
static inline unsigned gblk(long long n) { return (unsigned)((n + 255) / 256); }
static inline unsigned cdivu(long long a, long long b) { return (unsigned)((a + b - 1) / b); }
static inline int ceil64(int k) { return (k + 63) & ~63; }

static void gemm(const unsigned short* A, const unsigned short* W, float* C,
                 int M, int Kp, int Npad, int beta, hipStream_t s) {
  dim3 g(Npad / 16, cdivu(M, 16 * 8 * GEMM_MT));   // M % 512 == 0 at all call sites
  k_gemm_wmma<<<g, 256, 0, s>>>(A, W, C, M, Kp, Npad, beta);
}

// KANLinear: C = silu(X)@base^T + bsplines(X)@(spline*scaler)^T
static void run_kan(const float* X, int Ss, int In, int Out, int Npad,
                    const float* wb, const float* sp, const float* sc,
                    float* C, unsigned short* SBF, unsigned short* ACT,
                    unsigned short* WBUF, int rows, hipStream_t s) {
  const int Kb = ceil64(In);
  k_pack_R<<<gblk((long long)Npad * Kb), 256, 0, s>>>(wb, In, Out, WBUF, Kb, Npad);
  k_cvt_bf16<<<gblk((long long)rows * Kb), 256, 0, s>>>(X, Ss, SBF, Kb, In, rows, 1);
  gemm(SBF, WBUF, C, rows, Kb, Npad, 0, s);
  k_pack_spline<<<gblk((long long)Npad * In * 12), 256, 0, s>>>(sp, sc, In, Out, WBUF, Npad);
  k_bspline_bf16<<<gblk((long long)rows * In), 256, 0, s>>>(X, Ss, In, ACT, rows);
  gemm(ACT, WBUF, C, rows, In * 12, Npad, 1, s);   // In*12 is a multiple of 64
}

extern "C" void kernel_launch(void* const* d_in, const int* in_sizes, int n_in,
                              void* d_out, int out_size, void* d_ws, size_t ws_size,
                              hipStream_t stream) {
  (void)in_sizes; (void)n_in; (void)out_size; (void)ws_size;
  const int NN = 40960, EE = 163840, Bn = 1024, NPG = 40;
  const long long EF = (long long)EE + NN;

#define FP(i) ((const float*)d_in[i])
#define IP(i) ((const int*)d_in[i])
  // input indices in setup_inputs() dict order (nested dicts in insertion order)
  const float *X = FP(0);
  const float *W1 = FP(1), *AS1 = FP(2), *AD1 = FP(3), *B1 = FP(4), *BN1G = FP(5), *BN1B = FP(6);
  const float *W2 = FP(7), *AS2 = FP(8), *AD2 = FP(9), *B2 = FP(10), *BN2G = FP(11), *BN2B = FP(12);
  const float *W3 = FP(13), *B3 = FP(14), *BN3G = FP(15), *BN3B = FP(16);
  const float *W4 = FP(17), *AS4 = FP(18), *AD4 = FP(19), *B4 = FP(20);
  const float *FG1B = FP(21), *FG1S = FP(22), *FG1C = FP(23);
  const float *EMB = FP(24);
  const float *C1W = FP(25), *C1B = FP(26), *BX1G = FP(27), *BX1B = FP(28);
  const float *C2W = FP(29), *C2B = FP(30), *BX2G = FP(31), *BX2B = FP(32);
  const float *XT1B = FP(33), *XT1S = FP(34), *XT1C = FP(35);
  const float *XT2B = FP(36), *XT2S = FP(37), *XT2C = FP(38);
  const float *XT3B = FP(39), *XT3S = FP(40), *XT3C = FP(41);
  const float *XT4B = FP(42), *XT4S = FP(43), *XT4C = FP(44);
  const float *F1B = FP(45), *F1S = FP(46), *F1C = FP(47), *BF1G = FP(48), *BF1B = FP(49);
  const float *F2B = FP(50), *F2S = FP(51), *F2C = FP(52), *BF2G = FP(53), *BF2B = FP(54);
  const float *F3B = FP(55), *F3S = FP(56), *F3C = FP(57);
  const float *OB = FP(58), *OS = FP(59), *OC = FP(60);
  const int *EI = IP(61), *TGT = IP(63);

  // deterministic bump allocator over d_ws
  size_t off = 0;
  char* wsb = (char*)d_ws;
  auto carve = [&](size_t bytes) -> char* {
    char* p = wsb + off;
    off += (bytes + 255) & ~(size_t)255;
    return p;
  };
  float*          BIG0 = (float*)carve((size_t)NN * 1024 * 4);            // GEMM out (xl/xw)
  float*          BIG1 = (float*)carve((size_t)NN * 1024 * 4);            // aggregated h
  unsigned short* ACT  = (unsigned short*)carve((size_t)1024 * 46464 * 2);// bf16 activations / bsplines
  unsigned short* SBF  = (unsigned short*)carve((size_t)1024 * 3904 * 2); // silu bf16 (KAN)
  unsigned short* WBUF = (unsigned short*)carve((size_t)256 * 46464 * 2); // packed weights bf16
  float*    ssrc = (float*)carve((size_t)NN * 10 * 4);
  float*    sdst = (float*)carve((size_t)NN * 10 * 4);
  unsigned* mbuf = (unsigned*)carve((size_t)NN * 10 * 4);
  float*    sumb = (float*)carve((size_t)NN * 10 * 4);
  float*    ew   = (float*)carve((size_t)EF * 10 * 4);
  float*    deg  = (float*)carve((size_t)NN * 4);
  float*    gpool= (float*)carve((size_t)Bn * 128 * 4);
  float*    xg   = (float*)carve((size_t)Bn * 128 * 4);
  float*    embx = (float*)carve((size_t)Bn * 2048 * 4);
  float*    convb= (float*)carve((size_t)Bn * 3872 * 4);
  float*    xtA  = (float*)carve((size_t)Bn * 128 * 4);
  float*    xtB  = (float*)carve((size_t)Bn * 256 * 4);
  float*    xt384= (float*)carve((size_t)Bn * 384 * 4);
  float*    xtC  = (float*)carve((size_t)Bn * 256 * 4);
  float*    xtD  = (float*)carve((size_t)Bn * 128 * 4);
  float*    xcC  = (float*)carve((size_t)Bn * 256 * 4);
  float*    t1024= (float*)carve((size_t)Bn * 1024 * 4);
  float*    t256 = (float*)carve((size_t)Bn * 256 * 4);
  float*    t256b= (float*)carve((size_t)Bn * 256 * 4);
  float*    t16  = (float*)carve((size_t)Bn * 16 * 4);

  hipStream_t s = stream;

  // GAT layer runner (shared by layers 1, 2, 4)
  auto gat_layer = [&](int H, int C, int Ns, const float* as, const float* ad) {
    k_gat_scores<<<gblk((long long)NN * H), 256, 0, s>>>(BIG0, Ns, as, ssrc, NN, H, C);
    k_gat_scores<<<gblk((long long)NN * H), 256, 0, s>>>(BIG0, Ns, ad, sdst, NN, H, C);
    k_fill_u32<<<gblk((long long)NN * H), 256, 0, s>>>(mbuf, 0u, (long long)NN * H);
    k_fill_f32<<<gblk((long long)NN * H), 256, 0, s>>>(sumb, 0.f, (long long)NN * H);
    k_fill_f32<<<gblk((long long)NN * Ns), 256, 0, s>>>(BIG1, 0.f, (long long)NN * Ns);
    k_edge_max   <<<gblk(EF * H), 256, 0, s>>>(EI, EE, NN, H, ssrc, sdst, mbuf);
    k_edge_expsum<<<gblk(EF * H), 256, 0, s>>>(EI, EE, NN, H, ssrc, sdst, mbuf, ew, sumb);
    k_edge_aggr  <<<gblk(EF * H), 256, 0, s>>>(EI, EE, NN, H, C, Ns, BIG0, ew, sumb, BIG1);
  };

  // ---- GAT layer 1: 78 -> 780 (H=10,C=78), ELU, BN affine -------------------
  k_cvt_bf16<<<gblk((long long)NN * 128), 256, 0, s>>>(X, 78, ACT, 128, 78, NN, 0);
  k_pack_T<<<gblk((long long)784 * 128), 256, 0, s>>>(W1, 78, 780, WBUF, 128, 784);
  gemm(ACT, WBUF, BIG0, NN, 128, 784, 0, s);
  gat_layer(10, 78, 784, AS1, AD1);
  k_epilogue<<<gblk((long long)NN * 780), 256, 0, s>>>(BIG1, 784, NN, 780, B1, 1, BN1G, BN1B, 0);
  k_cvt_bf16<<<gblk((long long)NN * 832), 256, 0, s>>>(BIG1, 784, ACT, 832, 780, NN, 0);

  // ---- GAT layer 2: 780 -> 256 (H=1,C=256), BN affine -----------------------
  k_pack_T<<<gblk((long long)256 * 832), 256, 0, s>>>(W2, 780, 256, WBUF, 832, 256);
  gemm(ACT, WBUF, BIG0, NN, 832, 256, 0, s);
  gat_layer(1, 256, 256, AS2, AD2);
  k_epilogue<<<gblk((long long)NN * 256), 256, 0, s>>>(BIG1, 256, NN, 256, B2, 0, BN2G, BN2B, 0);
  k_cvt_bf16<<<gblk((long long)NN * 256), 256, 0, s>>>(BIG1, 256, ACT, 256, 256, NN, 0);

  // ---- GCN layer 3: 256 -> 1024, BN affine, ReLU ----------------------------
  k_pack_T<<<gblk((long long)1024 * 256), 256, 0, s>>>(W3, 256, 1024, WBUF, 256, 1024);
  gemm(ACT, WBUF, BIG0, NN, 256, 1024, 0, s);
  k_fill_f32<<<gblk((long long)NN), 256, 0, s>>>(deg, 0.f, NN);
  k_deg<<<gblk(EF), 256, 0, s>>>(EI, EE, NN, deg);
  k_fill_f32<<<gblk((long long)NN * 1024), 256, 0, s>>>(BIG1, 0.f, (long long)NN * 1024);
  k_gcn_aggr<<<gblk(EF * 32), 256, 0, s>>>(EI, EE, NN, deg, BIG0, 1024, BIG1, 1024, 32);
  k_epilogue<<<gblk((long long)NN * 1024), 256, 0, s>>>(BIG1, 1024, NN, 1024, B3, 0, BN3G, BN3B, 1);
  k_cvt_bf16<<<gblk((long long)NN * 1024), 256, 0, s>>>(BIG1, 1024, ACT, 1024, 1024, NN, 0);

  // ---- GAT layer 4: 1024 -> 128 (H=1,C=128), ReLU ---------------------------
  k_pack_T<<<gblk((long long)128 * 1024), 256, 0, s>>>(W4, 1024, 128, WBUF, 1024, 128);
  gemm(ACT, WBUF, BIG0, NN, 1024, 128, 0, s);
  gat_layer(1, 128, 128, AS4, AD4);
  k_epilogue<<<gblk((long long)NN * 128), 256, 0, s>>>(BIG1, 128, NN, 128, B4, 0, nullptr, nullptr, 1);

  // ---- global max pool + KAN fc_g1 + ReLU -----------------------------------
  k_pool_max<<<gblk((long long)Bn * 128), 256, 0, s>>>(BIG1, 128, gpool, Bn, 128, NPG);
  run_kan(gpool, 128, 128, 128, 128, FG1B, FG1S, FG1C, xg, SBF, ACT, WBUF, Bn, s);
  k_epilogue<<<gblk((long long)Bn * 128), 256, 0, s>>>(xg, 128, Bn, 128, nullptr, 0, nullptr, nullptr, 1);

  // ---- token branch: embedding + two conv/KAN paths -------------------------
  k_emb<<<gblk((long long)Bn * 2048), 256, 0, s>>>(EMB, TGT, embx, Bn);
  k_conv<<<gblk((long long)Bn * 32 * 121), 256, 0, s>>>(embx, C1W, C1B, BX1G, BX1B, convb, Bn);
  run_kan(convb, 3872, 3872, 128, 128, XT1B, XT1S, XT1C, xtA, SBF, ACT, WBUF, Bn, s);
  k_epilogue<<<gblk((long long)Bn * 128), 256, 0, s>>>(xtA, 128, Bn, 128, nullptr, 0, nullptr, nullptr, 1);
  k_conv<<<gblk((long long)Bn * 32 * 121), 256, 0, s>>>(embx, C2W, C2B, BX2G, BX2B, convb, Bn);
  run_kan(convb, 3872, 3872, 256, 256, XT2B, XT2S, XT2C, xtB, SBF, ACT, WBUF, Bn, s);
  k_epilogue<<<gblk((long long)Bn * 256), 256, 0, s>>>(xtB, 256, Bn, 256, nullptr, 0, nullptr, nullptr, 1);

  k_copycols<<<gblk((long long)Bn * 128), 256, 0, s>>>(xtA, 128, xt384, 384, 0, 128, Bn);
  k_copycols<<<gblk((long long)Bn * 256), 256, 0, s>>>(xtB, 256, xt384, 384, 128, 256, Bn);
  run_kan(xt384, 384, 384, 256, 256, XT3B, XT3S, XT3C, xtC, SBF, ACT, WBUF, Bn, s);
  k_epilogue<<<gblk((long long)Bn * 256), 256, 0, s>>>(xtC, 256, Bn, 256, nullptr, 0, nullptr, nullptr, 1);
  run_kan(xtC, 256, 256, 128, 128, XT4B, XT4S, XT4C, xtD, SBF, ACT, WBUF, Bn, s);

  // ---- head: concat + KAN fc1/fc2/fc3/out -----------------------------------
  k_copycols<<<gblk((long long)Bn * 128), 256, 0, s>>>(xg,  128, xcC, 256, 0,   128, Bn);
  k_copycols<<<gblk((long long)Bn * 128), 256, 0, s>>>(xtD, 128, xcC, 256, 128, 128, Bn);
  run_kan(xcC, 256, 256, 1024, 1024, F1B, F1S, F1C, t1024, SBF, ACT, WBUF, Bn, s);
  k_epilogue<<<gblk((long long)Bn * 1024), 256, 0, s>>>(t1024, 1024, Bn, 1024, nullptr, 0, BF1G, BF1B, 1);
  run_kan(t1024, 1024, 1024, 256, 256, F2B, F2S, F2C, t256, SBF, ACT, WBUF, Bn, s);
  k_epilogue<<<gblk((long long)Bn * 256), 256, 0, s>>>(t256, 256, Bn, 256, nullptr, 0, BF2G, BF2B, 1);
  run_kan(t256, 256, 256, 256, 256, F3B, F3S, F3C, t256b, SBF, ACT, WBUF, Bn, s);
  run_kan(t256b, 256, 256, 1, 16, OB, OS, OC, t16, SBF, ACT, WBUF, Bn, s);
  k_copycols<<<gblk((long long)Bn), 256, 0, s>>>(t16, 16, (float*)d_out, 1, 0, 1, Bn);
}
